// QLSTMLayer_10247791968828
// MI455X (gfx1250) — matrix operand order repaired and checked by reading, not compiled, not run on hardware
//
#include <hip/hip_runtime.h>
#include <hip/hip_bf16.h>

// ---------------- problem constants ----------------
#define BATCH   4
#define SEQ     4096
#define DIM     1024
#define ROWS    (BATCH * SEQ)          // 16384
#define NTILE   (DIM / 16)             // 64 N-tiles
#define NSEG    32
#define SEGLEN  (SEQ / NSEG)           // 128

typedef __attribute__((ext_vector_type(16))) __bf16 v16bf;
typedef __attribute__((ext_vector_type(8)))  __bf16 v8bf;
typedef __attribute__((ext_vector_type(8)))  float  v8f;

union V16U { v16bf v; v8bf h[2]; };

__device__ __forceinline__ float fsig(float x) {
    x = fminf(fmaxf(x, -30.0f), 30.0f);
    return 1.0f / (1.0f + __expf(-x));
}
__device__ __forceinline__ float ftanh(float x) {
    x = fminf(fmaxf(x, -15.0f), 15.0f);
    float e2 = __expf(2.0f * x);
    return (e2 - 1.0f) / (e2 + 1.0f);
}

// ---------------- 1) f32 -> bf16 conversion ----------------
__global__ void cvt_f32_bf16(const float* __restrict__ src,
                             __bf16* __restrict__ dst, int n) {
    int i = blockIdx.x * blockDim.x + threadIdx.x;
    int stride = gridDim.x * blockDim.x;
    for (; i < n; i += stride) dst[i] = (__bf16)src[i];
}

// ---------------- 2) fused 4-gate GEMM + nonlinearities ----------------
// One wave per 32x16 output tile (2 M-tiles x 4 gates = 8 WMMA accumulators).
// A = x tile (M=rows, K=d); B lane-n = row (e0+n) of W => contiguous 32B
// loads, no transpose needed. B fragments reused across both M-tiles,
// A fragments reused across all 4 gates: 12 b128 loads per 8 WMMAs.
__global__ __launch_bounds__(256) void gemm_gates(
    const __bf16* __restrict__ xb,   // [ROWS, DIM] bf16
    const __bf16* __restrict__ wb,   // 4 x [DIM, DIM] bf16 (Wf,Wi,Wig,Wog)
    const float* __restrict__ bfv, const float* __restrict__ biv,
    const float* __restrict__ bigv, const float* __restrict__ bogv,
    float* __restrict__ F,           // forget gate, f32
    float* __restrict__ I,           // input term, f32 (staged in d_out y region)
    __bf16* __restrict__ OG)         // output-gate sigmoid, bf16
{
    const int gw   = blockIdx.x * 8 + (threadIdx.x >> 5);   // global wave id
    const int lane = threadIdx.x & 31;
    const int mt2 = gw >> 6;                // 0..511  (pair of 16-row tiles)
    const int nt  = gw & (NTILE - 1);       // 0..63   (channel tile)
    const int r0 = mt2 << 5;                // 32 rows per wave
    const int e0 = nt << 4;
    const int half = lane >> 4;             // 0: lanes 0-15, 1: lanes 16-31
    const int l15  = lane & 15;

    // A fragment bases: row r0(+16)+l15; K 0-15 (half=0) / 16-31 (half=1)
    const __bf16* arow0 = xb + (size_t)(r0 + l15) * DIM + half * 16;
    const __bf16* arow1 = arow0 + (size_t)16 * DIM;
    // B fragment base: W row e0+l15; K 0-15 (half=0) / 16-31 (half=1)
    const __bf16* brow = wb + (size_t)(e0 + l15) * DIM + half * 16;
    const size_t WS = (size_t)DIM * DIM;

    v8f acc[2][4] = {};
    #pragma unroll 2
    for (int kk = 0; kk < DIM; kk += 32) {
        V16U a0, a1;
        a0.h[0] = *(const v8bf*)(arow0 + kk);
        a0.h[1] = *(const v8bf*)(arow0 + kk + 8);
        a1.h[0] = *(const v8bf*)(arow1 + kk);
        a1.h[1] = *(const v8bf*)(arow1 + kk + 8);
        v16bf b0 = *(const v16bf*)(brow + kk);            // Wf
        v16bf b1 = *(const v16bf*)(brow + WS + kk);       // Wi
        v16bf b2 = *(const v16bf*)(brow + 2 * WS + kk);   // Wig
        v16bf b3 = *(const v16bf*)(brow + 3 * WS + kk);   // Wog
        acc[0][0] = __builtin_amdgcn_wmma_f32_16x16x32_bf16(false, a0.v, false, b0,
                                                            (short)0, acc[0][0], false, false);
        acc[1][0] = __builtin_amdgcn_wmma_f32_16x16x32_bf16(false, a1.v, false, b0,
                                                            (short)0, acc[1][0], false, false);
        acc[0][1] = __builtin_amdgcn_wmma_f32_16x16x32_bf16(false, a0.v, false, b1,
                                                            (short)0, acc[0][1], false, false);
        acc[1][1] = __builtin_amdgcn_wmma_f32_16x16x32_bf16(false, a1.v, false, b1,
                                                            (short)0, acc[1][1], false, false);
        acc[0][2] = __builtin_amdgcn_wmma_f32_16x16x32_bf16(false, a0.v, false, b2,
                                                            (short)0, acc[0][2], false, false);
        acc[1][2] = __builtin_amdgcn_wmma_f32_16x16x32_bf16(false, a1.v, false, b2,
                                                            (short)0, acc[1][2], false, false);
        acc[0][3] = __builtin_amdgcn_wmma_f32_16x16x32_bf16(false, a0.v, false, b3,
                                                            (short)0, acc[0][3], false, false);
        acc[1][3] = __builtin_amdgcn_wmma_f32_16x16x32_bf16(false, a1.v, false, b3,
                                                            (short)0, acc[1][3], false, false);
    }

    // Epilogue. C/D layout: lanes 0-15 -> N=lane, M=vgpr r; lanes 16-31 -> M=8+r.
    const int e = e0 + l15;
    const float cf = bfv[e], ci = biv[e], cg = bigv[e], co = bogv[e];
    #pragma unroll
    for (int m = 0; m < 2; ++m) {
        const int rowb = r0 + m * 16 + half * 8;
        #pragma unroll
        for (int r = 0; r < 8; ++r) {
            const size_t idx = (size_t)(rowb + r) * DIM + e;
            float f  = fsig(acc[m][0][r] + cf);
            float it = ftanh(acc[m][1][r] + ci) * fsig(acc[m][2][r] + cg);
            float og = fsig(acc[m][3][r] + co);
            F[idx]  = f;
            I[idx]  = it;
            OG[idx] = (__bf16)og;
        }
    }
}

// ---------------- 3a) per-segment scan reduce: (A,B) per segment ----------------
__global__ __launch_bounds__(256) void seg_reduce(
    const float* __restrict__ F, const float* __restrict__ I,
    float* __restrict__ segA, float* __restrict__ segB)
{
    const int tid = blockIdx.x * blockDim.x + threadIdx.x;  // 0..131071
    const int e  = tid & (DIM - 1);
    const int bs = tid >> 10;            // b*NSEG + s
    const int b  = bs >> 5;
    const int s  = bs & (NSEG - 1);
    const int row0 = b * SEQ + s * SEGLEN;
    float a = 1.0f, h = 0.0f;
    for (int t = 0; t < SEGLEN; ++t) {
        const size_t idx = (size_t)(row0 + t) * DIM + e;
        float f = F[idx];
        a *= f;
        h = fmaf(f, h, I[idx]);
    }
    segA[tid] = a;
    segB[tid] = h;
}

// ---------------- 3b) sequential scan over 32 segments (tiny) ----------------
__global__ __launch_bounds__(256) void seg_scan(
    const float* __restrict__ hidden,
    const float* __restrict__ segA, const float* __restrict__ segB,
    float* __restrict__ hstart)
{
    const int tid = blockIdx.x * blockDim.x + threadIdx.x;  // 0..4095
    const int e = tid & (DIM - 1);
    const int b = tid >> 10;
    float h = hidden[tid];
    #pragma unroll
    for (int s = 0; s < NSEG; ++s) {
        const int j = (b * NSEG + s) * DIM + e;
        hstart[j] = h;
        h = fmaf(segA[j], h, segB[j]);
    }
}

// ---------------- 3c) apply: rescan segment, emit y and h_last ----------------
__global__ __launch_bounds__(256) void seg_apply(
    const float* __restrict__ F, const __bf16* __restrict__ OG,
    const float* __restrict__ hstart,
    float* __restrict__ Y,               // read i, overwrite with y (same buffer)
    float* __restrict__ hlast)
{
    const int tid = blockIdx.x * blockDim.x + threadIdx.x;  // 0..131071
    const int e  = tid & (DIM - 1);
    const int bs = tid >> 10;
    const int b  = bs >> 5;
    const int s  = bs & (NSEG - 1);
    const int row0 = b * SEQ + s * SEGLEN;
    float h = hstart[tid];
    for (int t = 0; t < SEGLEN; ++t) {
        const size_t idx = (size_t)(row0 + t) * DIM + e;
        float f  = F[idx];
        float it = Y[idx];
        float og = (float)OG[idx];
        h = fmaf(f, h, it);
        Y[idx] = ftanh(h) * og;
    }
    if (s == NSEG - 1) hlast[b * DIM + e] = h;
}

// ---------------- host launch ----------------
extern "C" void kernel_launch(void* const* d_in, const int* in_sizes, int n_in,
                              void* d_out, int out_size, void* d_ws, size_t ws_size,
                              hipStream_t stream) {
    const float* x    = (const float*)d_in[0];
    const float* hid  = (const float*)d_in[1];
    const float* Wf   = (const float*)d_in[2];
    const float* bfv  = (const float*)d_in[3];
    const float* Wi   = (const float*)d_in[4];
    const float* biv  = (const float*)d_in[5];
    const float* Wig  = (const float*)d_in[6];
    const float* bigv = (const float*)d_in[7];
    const float* Wog  = (const float*)d_in[8];
    const float* bogv = (const float*)d_in[9];

    float* y     = (float*)d_out;                        // [B, L, D]
    float* hlast = y + (size_t)ROWS * DIM;               // [B, D]

    // workspace layout (bytes)
    char* ws = (char*)d_ws;
    __bf16* xbf   = (__bf16*)(ws);                       //  33,554,432 B
    __bf16* wbf   = (__bf16*)(ws + 33554432);            //   8,388,608 B
    float*  Fbuf  = (float*) (ws + 41943040);            //  67,108,864 B
    __bf16* OGbuf = (__bf16*)(ws + 109051904);           //  33,554,432 B
    float*  segA  = (float*) (ws + 142606336);           //     524,288 B
    float*  segB  = (float*) (ws + 143130624);           //     524,288 B
    float*  hstrt = (float*) (ws + 143654912);           //     524,288 B
    (void)ws_size; (void)in_sizes; (void)n_in; (void)out_size;

    const size_t WS = (size_t)DIM * DIM;

    // 1) precision conversion (everything then lives in the 192MB L2)
    cvt_f32_bf16<<<8192, 256, 0, stream>>>(x, xbf, ROWS * DIM);
    cvt_f32_bf16<<<1024, 256, 0, stream>>>(Wf,  wbf + 0 * WS, DIM * DIM);
    cvt_f32_bf16<<<1024, 256, 0, stream>>>(Wi,  wbf + 1 * WS, DIM * DIM);
    cvt_f32_bf16<<<1024, 256, 0, stream>>>(Wig, wbf + 2 * WS, DIM * DIM);
    cvt_f32_bf16<<<1024, 256, 0, stream>>>(Wog, wbf + 3 * WS, DIM * DIM);

    // 2) fused 4-gate WMMA GEMM + bias + sigmoid/tanh
    //    waves = (ROWS/32)*(DIM/16) = 32768 -> 4096 blocks x 8 waves
    gemm_gates<<<4096, 256, 0, stream>>>(xbf, wbf, bfv, biv, bigv, bogv,
                                         Fbuf, /*I staged in y*/ y, OGbuf);

    // 3) chunked associative scan over L (32 segments of 128)
    seg_reduce<<<512, 256, 0, stream>>>(Fbuf, y, segA, segB);
    seg_scan<<<16, 256, 0, stream>>>(hid, segA, segB, hstrt);
    seg_apply<<<512, 256, 0, stream>>>(Fbuf, OGbuf, hstrt, y, hlast);
}
